// VectorQuantizer_22497038696783
// MI455X (gfx1250) — matrix operand order repaired and checked by reading, not compiled, not run on hardware
//
#include <hip/hip_runtime.h>

typedef __attribute__((ext_vector_type(16))) __bf16 v16bf;
typedef __attribute__((ext_vector_type(8)))  __bf16 v8bf;
typedef __attribute__((ext_vector_type(8)))  float  v8f;
typedef __attribute__((ext_vector_type(4)))  unsigned v4u;
typedef __attribute__((ext_vector_type(8)))  int      v8i;
typedef __attribute__((ext_vector_type(4)))  int      v4i;

#define N_ROWS   16384     // 4*4096
#define K_CODES  8192
#define D_DIM    256
#define LDS_STR  264       // 256 + 8 bf16 pad (TDM pad: 4 dwords per 128 dwords)
#define KSPLIT   8
#define KS_LEN   (K_CODES / KSPLIT)      // 1024 codes per split
#define KS_TILES (KS_LEN / 64)           // 16 tiles of 64 codes
#define MB_ROWS  256                     // rows per workgroup (8 waves x 32)
#define MB_CNT   (N_ROWS / MB_ROWS)      // 64 row-blocks
#define OUT_ELEMS (N_ROWS * D_DIM)

// f32 -> bf16 round-to-nearest-even (bit-level)
static __device__ __forceinline__ __bf16 bfc(float f) {
    unsigned u = __builtin_bit_cast(unsigned, f);
    u = u + 0x7FFFu + ((u >> 16) & 1u);
    return __builtin_bit_cast(__bf16, (unsigned short)(u >> 16));
}
static __device__ __forceinline__ unsigned short bfu(float f) {
    unsigned u = __builtin_bit_cast(unsigned, f);
    u = u + 0x7FFFu + ((u >> 16) & 1u);
    return (unsigned short)(u >> 16);
}

// ---------------------------------------------------------------------------
// TDM: DMA one 64x256 bf16 tile (row-major) into LDS with a 16-byte pad after
// every 512-byte row (LDS row stride = LDS_STR*2 = 528 B).  D# per ISA 8.3/8.4.
// ---------------------------------------------------------------------------
static __device__ __forceinline__ void tdm_load_tile(const unsigned short* gsrc,
                                                     unsigned lds_byte_addr) {
    unsigned long long ga = (unsigned long long)gsrc;
    v4u g0;
    g0[0] = 1u;                                           // count=1, user D#
    g0[1] = lds_byte_addr;                                // LDS dest (bytes)
    g0[2] = (unsigned)(ga & 0xFFFFFFFFu);                 // global_addr[31:0]
    g0[3] = (unsigned)((ga >> 32) & 0x01FFFFFFu)          // global_addr[56:32]
            | 0x80000000u;                                // type=2 ("image")
    v8i g1;
    g1[0] = (int)((1u << 16)      // data_size = 2 bytes
                | (1u << 20)      // pad_enable
                | (6u << 22)      // pad_interval: 2^(6+1)=128 dwords (=512 B row)
                | (3u << 25));    // pad_amount: 3+1 = 4 dwords (=16 B)
    g1[1] = (int)(256u << 16);    // tensor_dim0 = 256 (bits 63:48)
    g1[2] = (int)(64u << 16);     // tensor_dim0 hi=0; tensor_dim1 = 64 (bits 95:80)
    g1[3] = (int)(256u << 16);    // tensor_dim1 hi=0; tile_dim0 = 256 (bits 127:112)
    g1[4] = (int)64u;             // tile_dim1 = 64; tile_dim2 = 0
    g1[5] = (int)256u;            // tensor_dim0_stride = 256 (bits 207:160, lo32)
    g1[6] = 0;                    // dim0_stride hi; tensor_dim1_stride lo
    g1[7] = 0;                    // tensor_dim1_stride hi
    v4i zz = {0, 0, 0, 0};        // groups 2/3: unused (2-D tile)
#if defined(__clang_major__) && (__clang_major__ >= 23)
    v8i z8 = {0, 0, 0, 0, 0, 0, 0, 0};
    __builtin_amdgcn_tensor_load_to_lds(g0, g1, zz, zz, z8, 0);
#else
    __builtin_amdgcn_tensor_load_to_lds(g0, g1, zz, zz, 0);
#endif
}

// ---------------------------------------------------------------------------
// Kernel 1: e2h[k] = 0.5*||e_k||^2  and  e16 = bf16(e)  (row-major copy)
// ---------------------------------------------------------------------------
__global__ __launch_bounds__(256) void vq_prep(const float* __restrict__ e,
                                               float* __restrict__ e2h,
                                               unsigned short* __restrict__ e16) {
    int gt  = blockIdx.x * 256 + threadIdx.x;   // 0 .. 32767
    int r   = gt >> 2;
    int seg = (gt & 3) * 64;
    const float4* p = (const float4*)(e + (size_t)r * D_DIM + seg);
    uint4* q = (uint4*)(e16 + (size_t)r * D_DIM + seg);
    float s = 0.0f;
#pragma unroll
    for (int i = 0; i < 8; ++i) {
        float4 a = p[2 * i];
        float4 b = p[2 * i + 1];
        s += a.x * a.x + a.y * a.y + a.z * a.z + a.w * a.w;
        s += b.x * b.x + b.y * b.y + b.z * b.z + b.w * b.w;
        uint4 v;
        v.x = (unsigned)bfu(a.x) | ((unsigned)bfu(a.y) << 16);
        v.y = (unsigned)bfu(a.z) | ((unsigned)bfu(a.w) << 16);
        v.z = (unsigned)bfu(b.x) | ((unsigned)bfu(b.y) << 16);
        v.w = (unsigned)bfu(b.z) | ((unsigned)bfu(b.w) << 16);
        q[i] = v;
    }
    s += __shfl_xor(s, 1, 4);
    s += __shfl_xor(s, 2, 4);
    if ((gt & 3) == 0) e2h[r] = 0.5f * s;
}

// ---------------------------------------------------------------------------
// Kernel 2: K-split partial argmax_k (z.e_k - 0.5||e_k||^2) via WMMA bf16.
// 8 waves x 32 rows; codebook tiles arrive by TDM into double-buffered LDS:
// wave 0 issues tensor_load_to_lds for tile k+1, s_wait_tensorcnt gates tile k.
// ---------------------------------------------------------------------------
__global__ __launch_bounds__(256) void vq_argmax(const float* __restrict__ z,
                                                 const unsigned short* __restrict__ e16,
                                                 const float* __restrict__ e2h,
                                                 float* __restrict__ cand_val,
                                                 unsigned* __restrict__ cand_idx) {
    __shared__ __align__(16) unsigned short lds_e[2][64 * LDS_STR];   // 66 KB

    const int t     = threadIdx.x;
    const int lane  = t & 31;
    const int w     = t >> 5;
    const int m0    = blockIdx.x * MB_ROWS + w * 32;   // this wave's 32 rows
    const int ks0   = blockIdx.y * KS_LEN;             // this split's first code
    const int lrow  = lane & 15;
    const int hi    = lane >> 4;
    const int kbase = hi * 8;

    const unsigned lds0 = (unsigned)(unsigned long long)(void*)&lds_e[0][0];
    const unsigned lds1 = (unsigned)(unsigned long long)(void*)&lds_e[1][0];

    // Kick off DMA of tile 0 before the A-fragment setup.
    if (w == 0) tdm_load_tile(e16 + (size_t)ks0 * D_DIM, lds0);

    // ---- A fragments: two 16x256 tiles resident in 16 x v16bf ----
    v16bf A[2][8];
#pragma unroll
    for (int mt = 0; mt < 2; ++mt) {
        const float* zrow = z + (size_t)(m0 + mt * 16 + lrow) * D_DIM;
#pragma unroll
        for (int c = 0; c < 8; ++c) {
            const float* p = zrow + c * 32 + kbase;
            float4 f0 = *(const float4*)(p);
            float4 f1 = *(const float4*)(p + 4);
            float4 f2 = *(const float4*)(p + 16);
            float4 f3 = *(const float4*)(p + 20);
            v16bf a;
            a[0]  = bfc(f0.x); a[1]  = bfc(f0.y); a[2]  = bfc(f0.z); a[3]  = bfc(f0.w);
            a[4]  = bfc(f1.x); a[5]  = bfc(f1.y); a[6]  = bfc(f1.z); a[7]  = bfc(f1.w);
            a[8]  = bfc(f2.x); a[9]  = bfc(f2.y); a[10] = bfc(f2.z); a[11] = bfc(f2.w);
            a[12] = bfc(f3.x); a[13] = bfc(f3.y); a[14] = bfc(f3.z); a[15] = bfc(f3.w);
            A[mt][c] = a;
        }
    }

    float    bestv[2][8];
    unsigned besti[2][8];
#pragma unroll
    for (int mt = 0; mt < 2; ++mt)
#pragma unroll
        for (int v = 0; v < 8; ++v) { bestv[mt][v] = -3.0e38f; besti[mt][v] = 0u; }

    for (int kt = 0; kt < KS_TILES; ++kt) {
        const int k0 = ks0 + kt * 64;
        // Publish tile kt: wave 0 waits for its DMA, barrier releases everyone.
        if (w == 0) __builtin_amdgcn_s_wait_tensorcnt(0);
        __syncthreads();
        // Everyone is past reading buf[1-cur] (finished last iteration), so
        // its refill for tile kt+1 can start under this tile's compute.
        if (w == 0 && kt + 1 < KS_TILES)
            tdm_load_tile(e16 + (size_t)(k0 + 64) * D_DIM, (kt & 1) ? lds0 : lds1);

        const unsigned short* lb = &lds_e[kt & 1][0];
        auto loadB = [&](int nt, int c) -> v16bf {
            const unsigned short* bp = lb + (nt * 16 + lrow) * LDS_STR + c * 32 + hi * 8;
            v8bf lo = *(const v8bf*)bp;
            v8bf hh = *(const v8bf*)(bp + 16);
            return __builtin_shufflevector(lo, hh,
                   0, 1, 2, 3, 4, 5, 6, 7, 8, 9, 10, 11, 12, 13, 14, 15);
        };

#pragma unroll
        for (int nt = 0; nt < 4; ++nt) {
            const unsigned kcol = (unsigned)(k0 + nt * 16 + lrow);
            const float nb = -e2h[kcol];
            v8f acc0, acc1;
#pragma unroll
            for (int v = 0; v < 8; ++v) { acc0[v] = nb; acc1[v] = nb; }

            v16bf B = loadB(nt, 0);
#pragma unroll
            for (int c = 0; c < 8; ++c) {
                v16bf Bn = (c < 7) ? loadB(nt, c + 1) : B;   // prefetch next fragment
                acc0 = __builtin_amdgcn_wmma_f32_16x16x32_bf16(
                           false, A[0][c], false, B, (short)0, acc0, false, false);
                acc1 = __builtin_amdgcn_wmma_f32_16x16x32_bf16(
                           false, A[1][c], false, B, (short)0, acc1, false, false);
                B = Bn;
            }
#pragma unroll
            for (int v = 0; v < 8; ++v) {
                if (acc0[v] > bestv[0][v]) { bestv[0][v] = acc0[v]; besti[0][v] = kcol; }
                if (acc1[v] > bestv[1][v]) { bestv[1][v] = acc1[v]; besti[1][v] = kcol; }
            }
        }
    }

    // ---- argmax across the 16 lanes holding the same row (prefer min index) ----
    float* cv = cand_val + (size_t)blockIdx.y * N_ROWS;
    unsigned* ci = cand_idx + (size_t)blockIdx.y * N_ROWS;
#pragma unroll
    for (int mt = 0; mt < 2; ++mt) {
#pragma unroll
        for (int v = 0; v < 8; ++v) {
            float    bv = bestv[mt][v];
            unsigned bi = besti[mt][v];
#pragma unroll
            for (int m = 8; m >= 1; m >>= 1) {
                float    ov = __shfl_xor(bv, m, 16);
                unsigned oi = __shfl_xor(bi, m, 16);
                if (ov > bv || (ov == bv && oi < bi)) { bv = ov; bi = oi; }
            }
            if (lrow == 0) {
                const int row = m0 + mt * 16 + kbase + v;
                cv[row] = bv;
                ci[row] = bi;
            }
        }
    }
}

// ---------------------------------------------------------------------------
// Kernel 2b: merge the KSPLIT per-row candidates into the final index.
// ---------------------------------------------------------------------------
__global__ __launch_bounds__(256) void vq_merge(const float* __restrict__ cand_val,
                                                const unsigned* __restrict__ cand_idx,
                                                unsigned* __restrict__ idx_out) {
    const int row = blockIdx.x * 256 + threadIdx.x;
    float    bv = cand_val[row];
    unsigned bi = cand_idx[row];
#pragma unroll
    for (int s = 1; s < KSPLIT; ++s) {
        float    v = cand_val[(size_t)s * N_ROWS + row];
        unsigned i = cand_idx[(size_t)s * N_ROWS + row];
        if (v > bv || (v == bv && i < bi)) { bv = v; bi = i; }
    }
    idx_out[row] = bi;
}

// ---------------------------------------------------------------------------
// Kernel 3: exact f32 gather z_q = e[idx], write out, accumulate (z_q - z)^2
// ---------------------------------------------------------------------------
__global__ __launch_bounds__(256) void vq_gather(const float* __restrict__ z,
                                                 const float* __restrict__ e,
                                                 const unsigned* __restrict__ idx,
                                                 float* __restrict__ out,
                                                 float* __restrict__ partial) {
    const int t = threadIdx.x;
    const int g = blockIdx.x * 256 + t;      // float4 index, 0 .. 1048575
    const int row = g >> 6;
    const int c4  = (g & 63) * 4;
    const unsigned k = idx[row];
    const float4 q  = *(const float4*)(e + (size_t)k * D_DIM + c4);
    const float4 zz = *(const float4*)(z + (size_t)row * D_DIM + c4);
    ((float4*)out)[g] = q;
    float dx = q.x - zz.x, dy = q.y - zz.y, dz = q.z - zz.z, dw = q.w - zz.w;
    float s = dx * dx + dy * dy + dz * dz + dw * dw;

    s += __shfl_xor(s, 1);  s += __shfl_xor(s, 2);  s += __shfl_xor(s, 4);
    s += __shfl_xor(s, 8);  s += __shfl_xor(s, 16);
    __shared__ float red[8];
    if ((t & 31) == 0) red[t >> 5] = s;
    __syncthreads();
    if (t == 0) {
        float tot = 0.0f;
#pragma unroll
        for (int i = 0; i < 8; ++i) tot += red[i];
        partial[blockIdx.x] = tot;
    }
}

// ---------------------------------------------------------------------------
// Kernel 4: loss = 1.25 * sum / (N*D)
// ---------------------------------------------------------------------------
__global__ __launch_bounds__(256) void vq_loss(const float* __restrict__ partial,
                                               float* __restrict__ loss_out) {
    const int t = threadIdx.x;
    float s = 0.0f;
    for (int i = t; i < 4096; i += 256) s += partial[i];
    s += __shfl_xor(s, 1);  s += __shfl_xor(s, 2);  s += __shfl_xor(s, 4);
    s += __shfl_xor(s, 8);  s += __shfl_xor(s, 16);
    __shared__ float red[8];
    if ((t & 31) == 0) red[t >> 5] = s;
    __syncthreads();
    if (t == 0) {
        float tot = 0.0f;
#pragma unroll
        for (int i = 0; i < 8; ++i) tot += red[i];
        loss_out[0] = tot * (1.25f / (float)OUT_ELEMS);
    }
}

extern "C" void kernel_launch(void* const* d_in, const int* in_sizes, int n_in,
                              void* d_out, int out_size, void* d_ws, size_t ws_size,
                              hipStream_t stream) {
    const float* z = (const float*)d_in[0];   // [16384, 256] f32
    const float* e = (const float*)d_in[1];   // [8192, 256]  f32
    float* out = (float*)d_out;               // [4194304] z_q  +  [1] loss

    float*          e2h     = (float*)d_ws;                        // 8192 f32
    unsigned short* e16     = (unsigned short*)(e2h + K_CODES);    // 8192*256 bf16
    float*          cand_v  = (float*)(e16 + (size_t)K_CODES * D_DIM);
    unsigned*       cand_i  = (unsigned*)(cand_v + KSPLIT * N_ROWS);
    unsigned*       idxs    = cand_i + KSPLIT * N_ROWS;            // 16384 u32
    float*          partial = (float*)(idxs + N_ROWS);             // 4096 f32

    vq_prep  <<<128, 256, 0, stream>>>(e, e2h, e16);
    vq_argmax<<<dim3(MB_CNT, KSPLIT), 256, 0, stream>>>(z, e16, e2h, cand_v, cand_i);
    vq_merge <<<N_ROWS / 256, 256, 0, stream>>>(cand_v, cand_i, idxs);
    vq_gather<<<4096, 256, 0, stream>>>(z, e, idxs, out, partial);
    vq_loss  <<<1, 256, 0, stream>>>(partial, out + OUT_ELEMS);
}
